// MultiGranularityVSEncoder_47347719471325
// MI455X (gfx1250) — matrix operand order repaired and checked by reading, not compiled, not run on hardware
//
#include <hip/hip_runtime.h>

// ---------------------------------------------------------------------------
// MultiGranularityVSEncoder forward for MI455X (gfx1250).
// bf16 WMMA GEMM pipeline, Tensor-Data-Mover LDS staging (double buffered).
// B=64, N=512 nodes, D=512. ~86 GFLOP, ~200MB intermediates (fits 192MB L2).
// All GEMMs are canonicalized to  C = A * B^T  (A:[M,K], B:[N,K] row-major).
// ---------------------------------------------------------------------------

typedef __attribute__((ext_vector_type(16))) __bf16 v16bf;
typedef __attribute__((ext_vector_type(8)))  float  v8f;
typedef __attribute__((ext_vector_type(4)))  unsigned int v4u;
typedef __attribute__((ext_vector_type(8)))  int v8i;
typedef __attribute__((ext_vector_type(4)))  int v4i;

#if defined(__AMDGCN__) && defined(__has_builtin)
#if __has_builtin(__builtin_amdgcn_tensor_load_to_lds) && __has_builtin(__builtin_amdgcn_s_wait_tensorcnt)
#define USE_TDM 1
#endif
#endif
#ifndef USE_TDM
#define USE_TDM 0
#endif

#define BATCH 64
#define NNODE 512
#define DDIM  512
#define MTOT  (BATCH * NNODE)   // 32768

#define TM 128
#define TN 128
#define TK 32
#define ASTR 40                 // LDS row pitch in ushorts: 32 data + 8 pad (80B)

__device__ __forceinline__ unsigned short f2bf(float f) {
  unsigned int u = __float_as_uint(f);
  u += 0x7FFFu + ((u >> 16) & 1u);          // round-to-nearest-even
  return (unsigned short)(u >> 16);
}

#if USE_TDM
// LDS byte offset of a (generic) pointer into static LDS.
__device__ __forceinline__ unsigned lds_off(const void* p) {
  return (unsigned)(unsigned long long)(__attribute__((address_space(3))) const void*)p;
}

// Issue one TDM 2-D tile load: rows x 32 bf16 elements, global row pitch `ld`
// (elements), into LDS at byte offset `lds_byte` with HW padding of 4 DWORDs
// after every 16 DWORDs (row pitch 64B data + 16B pad = ASTR*2 bytes).
__device__ __forceinline__ void tdm_load_tile(unsigned lds_byte, const unsigned short* gsrc,
                                              unsigned rows, unsigned ld) {
  unsigned long long ga = (unsigned long long)(size_t)gsrc;
  v4u g0 = { 1u,                                   // count=1 (valid user D#)
             lds_byte,                             // lds_addr
             (unsigned)ga,                         // global_addr[31:0]
             (unsigned)((ga >> 32) & 0x01FFFFFFu) | (2u << 30) };  // addr[56:32] | type=2
  v8i g1 = { (int)((1u << 16)                      // data_size = 2 bytes
                 | (1u << 20)                      // pad_enable
                 | (3u << 22)                      // pad_interval: 16 DWORDs (64B)
                 | (3u << 25)),                    // pad_amount: 4 DWORDs (16B)
             (int)((unsigned)TK << 16),            // tensor_dim0 = 32 (lo16 in hi half)
             (int)(rows << 16),                    // tensor_dim1 lo16
             (int)((unsigned)TK << 16),            // tile_dim0 = 32
             (int)rows,                            // tile_dim1
             (int)ld,                              // tensor_dim0_stride lo32
             0, 0 };
  v4i z4 = { 0, 0, 0, 0 };
  v8i z8 = { 0, 0, 0, 0, 0, 0, 0, 0 };
  __builtin_amdgcn_tensor_load_to_lds(g0, g1, z4, z4, z8, 0);
}
#endif

// ---------------------------------------------------------------------------
// X = src + src * node_embed[node_type]  (concat of 4 segments) -> bf16
// ---------------------------------------------------------------------------
__global__ void build_x_kernel(const float* __restrict__ vo, const float* __restrict__ vi,
                               const float* __restrict__ sr, const float* __restrict__ ss,
                               const float* __restrict__ ntab, unsigned short* __restrict__ X) {
  int idx = blockIdx.x * blockDim.x + threadIdx.x;
  if (idx >= BATCH * NNODE * DDIM) return;
  int d = idx & (DDIM - 1);
  int n = (idx >> 9) & (NNODE - 1);
  int b = idx >> 18;
  int seg = n >> 7;
  int nl  = n & 127;
  const float* src = (seg == 0) ? vo : (seg == 1) ? vi : (seg == 2) ? sr : ss;
  float v = src[((size_t)b * 128 + nl) * DDIM + d];
  float g = 1.0f + ntab[seg * DDIM + d];
  X[idx] = f2bf(v * g);
}

// W (f32, [K,N] row-major) -> WT (bf16, [N,K] row-major), one-time 512x512
__global__ void transpose_f32_to_bf16_kernel(const float* __restrict__ w,
                                             unsigned short* __restrict__ wt) {
  int idx = blockIdx.x * blockDim.x + threadIdx.x;    // [0, 512*512)
  int n = idx & (DDIM - 1);
  int k = idx >> 9;
  wt[(size_t)n * DDIM + k] = f2bf(w[(size_t)k * DDIM + n]);
}

// ---------------------------------------------------------------------------
// C[M,N] = A[M,K] * B[N,K]^T  (+bias over n, relu, optional transposed store)
// 256 threads = 8 waves (4x2), block tile 128x128, wave tile 32x64 (2x4 WMMA
// accs), K-step 32.  TDM double-buffered LDS staging on gfx1250.
// ---------------------------------------------------------------------------
union FragBF { v16bf v; unsigned int u[8]; };

__global__ void __launch_bounds__(256)
wmma_gemm_kernel(const unsigned short* __restrict__ A, int lda, long long strideA,
                 const unsigned short* __restrict__ B, int ldb, long long strideB,
                 const float* __restrict__ bias, int relu,
                 float* __restrict__ Cf, unsigned short* __restrict__ Cb,
                 long long strideC, int transC, int ldc, int K) {
  __shared__ __align__(16) unsigned short Alds[2][TM * ASTR];
  __shared__ __align__(16) unsigned short Blds[2][TN * ASTR];

  const int t    = threadIdx.x;
  const int lane = t & 31;
  const int wave = t >> 5;
  const int wm   = (wave & 3) * 32;       // wave row origin in block tile
  const int wn   = (wave >> 2) * 64;      // wave col origin in block tile
  const int lrow = lane & 15;
  const int khi  = lane >> 4;

  const int z  = blockIdx.z;
  const unsigned short* Ab = A + (size_t)z * strideA;
  const unsigned short* Bb = B + (size_t)z * strideB;
  const int m0 = blockIdx.y * TM;
  const int n0 = blockIdx.x * TN;

  v8f zacc = {};
  v8f acc[2][4] = {{zacc, zacc, zacc, zacc}, {zacc, zacc, zacc, zacc}};

  int cur = 0;
#if USE_TDM
  const unsigned a_lds0 = lds_off(&Alds[0][0]);
  const unsigned b_lds0 = lds_off(&Blds[0][0]);
  const unsigned abuf_bytes = TM * ASTR * 2;
  const unsigned bbuf_bytes = TN * ASTR * 2;
  if (t < 32) {   // wave 0 drives the Tensor Data Mover
    tdm_load_tile(a_lds0, Ab + (size_t)m0 * lda, TM, (unsigned)lda);
    tdm_load_tile(b_lds0, Bb + (size_t)n0 * ldb, TN, (unsigned)ldb);
    __builtin_amdgcn_s_wait_tensorcnt(0);
  }
  __syncthreads();
#endif

  for (int k0 = 0; k0 < K; k0 += TK) {
#if USE_TDM
    const bool have_next = (k0 + TK) < K;
    if (have_next && t < 32) {
      tdm_load_tile(a_lds0 + (cur ^ 1) * abuf_bytes, Ab + (size_t)m0 * lda + (k0 + TK), TM, (unsigned)lda);
      tdm_load_tile(b_lds0 + (cur ^ 1) * bbuf_bytes, Bb + (size_t)n0 * ldb + (k0 + TK), TN, (unsigned)ldb);
    }
#else
    // fallback: manual staging of both 128x32 tiles (identical shapes)
    {
      const int arow = t >> 2;          // 0..63
      const int acol = (t & 3) * 8;     // 0,8,16,24
      *(uint4*)&Alds[0][arow * ASTR + acol] =
          *(const uint4*)(Ab + (size_t)(m0 + arow) * lda + k0 + acol);
      *(uint4*)&Alds[0][(arow + 64) * ASTR + acol] =
          *(const uint4*)(Ab + (size_t)(m0 + arow + 64) * lda + k0 + acol);
      *(uint4*)&Blds[0][arow * ASTR + acol] =
          *(const uint4*)(Bb + (size_t)(n0 + arow) * ldb + k0 + acol);
      *(uint4*)&Blds[0][(arow + 64) * ASTR + acol] =
          *(const uint4*)(Bb + (size_t)(n0 + arow + 64) * ldb + k0 + acol);
      if (k0 + TK < K)
        __builtin_prefetch(Ab + (size_t)(m0 + arow) * lda + (k0 + TK) + acol, 0, 1);
    }
    __syncthreads();
#endif

    // ---- fragments (ISA 7.12.2 16-bit layouts) + 8 WMMAs ----
    FragBF a0, a1, b[4];
    {
      const unsigned int* ap0 = (const unsigned int*)&Alds[cur][(wm + lrow) * ASTR];
      const unsigned int* ap1 = (const unsigned int*)&Alds[cur][(wm + 16 + lrow) * ASTR];
      const int kh2 = khi * 4;  // A: lanes<16 K{0-7,16-23}; lanes>=16 K{8-15,24-31}
#pragma unroll
      for (int j = 0; j < 4; ++j) {
        a0.u[j]     = ap0[kh2 + j];
        a0.u[4 + j] = ap0[8 + kh2 + j];
        a1.u[j]     = ap1[kh2 + j];
        a1.u[4 + j] = ap1[8 + kh2 + j];
      }
      const int kb = khi * 4;   // B: lanes<16 K{0-7,16-23}; lanes>=16 K{8-15,24-31}
#pragma unroll
      for (int jj = 0; jj < 4; ++jj) {
        const unsigned int* bp = (const unsigned int*)&Blds[cur][(wn + jj * 16 + lrow) * ASTR];
#pragma unroll
        for (int q = 0; q < 8; ++q) b[jj].u[q] = bp[kb + (q & 3) + ((q >> 2) << 3)];
      }
    }
#pragma unroll
    for (int jj = 0; jj < 4; ++jj) {
      acc[0][jj] = __builtin_amdgcn_wmma_f32_16x16x32_bf16(false, a0.v, false, b[jj].v,
                                                           (short)0, acc[0][jj], false, false);
      acc[1][jj] = __builtin_amdgcn_wmma_f32_16x16x32_bf16(false, a1.v, false, b[jj].v,
                                                           (short)0, acc[1][jj], false, false);
    }

#if USE_TDM
    if (have_next && t < 32) __builtin_amdgcn_s_wait_tensorcnt(0);
    __syncthreads();
    cur ^= 1;
#else
    __syncthreads();
#endif
  }

  // ---- epilogue: bias + relu; f32 and/or bf16; optional transposed store ----
  float*          Cfb = Cf ? Cf + (size_t)z * strideC : nullptr;
  unsigned short* Cbb = Cb ? Cb + (size_t)z * strideC : nullptr;
#pragma unroll
  for (int i = 0; i < 2; ++i) {
#pragma unroll
    for (int jj = 0; jj < 4; ++jj) {
      const int nglob = n0 + wn + jj * 16 + lrow;
      const float badd = bias ? bias[nglob] : 0.0f;
#pragma unroll
      for (int r = 0; r < 8; ++r) {
        const int mglob = m0 + wm + i * 16 + r + 8 * khi;
        float v = acc[i][jj][r] + badd;
        if (relu) v = fmaxf(v, 0.0f);
        const size_t off = transC ? (size_t)nglob * ldc + mglob
                                  : (size_t)mglob * ldc + nglob;
        if (Cfb) Cfb[off] = v;
        if (Cbb) Cbb[off] = f2bf(v);
      }
    }
  }
}

// ---------------------------------------------------------------------------
// Row softmax over 512-wide rows of ADJ (f32) -> bf16
// ---------------------------------------------------------------------------
__global__ void softmax_rows_kernel(const float* __restrict__ adj,
                                    unsigned short* __restrict__ aout) {
  __shared__ float red[256];
  const int t = threadIdx.x;
  const float* p = adj + (size_t)blockIdx.x * NNODE;
  float v0 = p[t], v1 = p[t + 256];

  red[t] = fmaxf(v0, v1);
  __syncthreads();
  for (int s = 128; s > 0; s >>= 1) {
    if (t < s) red[t] = fmaxf(red[t], red[t + s]);
    __syncthreads();
  }
  const float mx = red[0];
  __syncthreads();

  float e0 = __expf(v0 - mx), e1 = __expf(v1 - mx);
  red[t] = e0 + e1;
  __syncthreads();
  for (int s = 128; s > 0; s >>= 1) {
    if (t < s) red[t] += red[t + s];
    __syncthreads();
  }
  const float inv = 1.0f / red[0];

  unsigned short* q = aout + (size_t)blockIdx.x * NNODE;
  q[t]       = f2bf(e0 * inv);
  q[t + 256] = f2bf(e1 * inv);
}

// out[b,d] = mean_n R[b,n,d]  (deterministic, no atomics)
__global__ void mean_rows_kernel(const float* __restrict__ R, float* __restrict__ out) {
  int idx = blockIdx.x * blockDim.x + threadIdx.x;
  int b = idx >> 9;
  int d = idx & (DDIM - 1);
  const float* p = R + (size_t)b * NNODE * DDIM + d;
  float s = 0.0f;
  for (int n = 0; n < NNODE; ++n) s += p[(size_t)n * DDIM];
  out[idx] = s * (1.0f / (float)NNODE);
}

// ---------------------------------------------------------------------------
extern "C" void kernel_launch(void* const* d_in, const int* in_sizes, int n_in,
                              void* d_out, int out_size, void* d_ws, size_t ws_size,
                              hipStream_t stream) {
  const float* vo   = (const float*)d_in[0];
  const float* vi   = (const float*)d_in[1];
  const float* sr   = (const float*)d_in[2];
  const float* ss   = (const float*)d_in[3];
  const float* ntab = (const float*)d_in[4];
  const float* w1   = (const float*)d_in[5];
  const float* b1   = (const float*)d_in[6];
  const float* w2   = (const float*)d_in[7];
  const float* b2   = (const float*)d_in[8];
  const float* gw   = (const float*)d_in[9];
  const float* gb   = (const float*)d_in[10];
  float* out = (float*)d_out;

  char* ws = (char*)d_ws;
  size_t o = 0;
  const size_t n_xd = (size_t)MTOT * DDIM;
  unsigned short* X   = (unsigned short*)(ws + o); o += n_xd * 2;
  unsigned short* H1  = (unsigned short*)(ws + o); o += n_xd * 2;
  unsigned short* H2  = (unsigned short*)(ws + o); o += n_xd * 2;
  unsigned short* ST  = (unsigned short*)(ws + o); o += n_xd * 2;   // support, transposed [b][d][m]
  float*          ADJ = (float*)(ws + o);          o += n_xd * 4;
  unsigned short* W1T = (unsigned short*)(ws + o); o += (size_t)DDIM * DDIM * 2;
  unsigned short* W2T = (unsigned short*)(ws + o); o += (size_t)DDIM * DDIM * 2;
  unsigned short* WgT = (unsigned short*)(ws + o); o += (size_t)DDIM * DDIM * 2;
  unsigned short* Abf = H1;   // reuse: H1 dead after GEMM2
  float*          R   = ADJ;  // reuse: ADJ dead after softmax

  const long long SB = (long long)NNODE * DDIM;

  // one-time weight transpose+convert: W[K,N] f32 -> WT[N,K] bf16
  transpose_f32_to_bf16_kernel<<<1024, 256, 0, stream>>>(w1, W1T);
  transpose_f32_to_bf16_kernel<<<1024, 256, 0, stream>>>(w2, W2T);
  transpose_f32_to_bf16_kernel<<<1024, 256, 0, stream>>>(gw, WgT);

  build_x_kernel<<<(BATCH * NNODE * DDIM) / 256, 256, 0, stream>>>(vo, vi, sr, ss, ntab, X);

  // H1 = relu(X @ W1 + b1) = relu(X . W1T^T + b1)
  wmma_gemm_kernel<<<dim3(NNODE / TN, MTOT / TM, 1), 256, 0, stream>>>(
      X, DDIM, 0, W1T, DDIM, 0, b1, 1, nullptr, H1, 0, /*transC=*/0, DDIM, DDIM);

  // H2 = relu(H1 @ W2 + b2)
  wmma_gemm_kernel<<<dim3(NNODE / TN, MTOT / TM, 1), 256, 0, stream>>>(
      H1, DDIM, 0, W2T, DDIM, 0, b2, 1, nullptr, H2, 0, 0, DDIM, DDIM);

  // ADJ[b] = H2[b] @ H2[b]^T
  wmma_gemm_kernel<<<dim3(NNODE / TN, NNODE / TM, BATCH), 256, 0, stream>>>(
      H2, DDIM, SB, H2, DDIM, SB, nullptr, 0, ADJ, nullptr, SB, 0, NNODE, DDIM);

  // A = row-softmax(ADJ) -> bf16
  softmax_rows_kernel<<<BATCH * NNODE, 256, 0, stream>>>(ADJ, Abf);

  // ST[b] = (X[b] @ Wg + bg)^T   (batched, stored transposed for final GEMM)
  wmma_gemm_kernel<<<dim3(NNODE / TN, NNODE / TM, BATCH), 256, 0, stream>>>(
      X, DDIM, SB, WgT, DDIM, 0, gb, 0, nullptr, ST, SB, /*transC=*/1, NNODE, DDIM);

  // R[b] = relu(A[b] @ S[b]) = relu(A[b] . ST[b]^T)
  wmma_gemm_kernel<<<dim3(NNODE / TN, NNODE / TM, BATCH), 256, 0, stream>>>(
      Abf, DDIM, SB, ST, DDIM, SB, nullptr, 1, R, nullptr, SB, 0, NNODE, DDIM);

  mean_rows_kernel<<<(BATCH * DDIM) / 256, 256, 0, stream>>>(R, out);

  (void)in_sizes; (void)n_in; (void)out_size; (void)ws_size;
}
